// EMAComplex_61357902790806
// MI455X (gfx1250) — matrix operand order repaired and checked by reading, not compiled, not run on hardware
//
#include <hip/hip_runtime.h>
#include <hip/hip_bf16.h>

// MI455X / gfx1250, wave32. Conv3x3 done as implicit GEMM on v_wmma_f32_16x16x32_bf16
// with hi/lo bf16 split (3-term) for ~fp32 accuracy, fp32 accumulation.
// a1-contraction folded into the B matrix (a1 = softmax(gnb) is data-independent),
// so each WMMA directly yields the attention-weighted conv in column 0.

typedef __attribute__((ext_vector_type(16))) __bf16 v16bf;
typedef __attribute__((ext_vector_type(8)))  float  v8f;

#define NN 64
#define CG 8
#define FF 256
#define TT 256
#define FTI (1.0f/65536.0f)

// workspace offsets (in floats); total ~1,055,760 floats (~4.03 MB) of d_ws
#define OFF_ROW  0          // [2][64][8][256] sums over t
#define OFF_COL  262144     // [2][64][8][256] sums over f
#define OFF_CORN 524288     // [2][64][8][4]   corner elements
#define OFF_WF   528384     // [2][64][8][256] f-gates
#define OFF_WT   790528     // [2][64][8][256] t-gates
#define OFF_A1   1052672    // [2][8]
#define OFF_A2   1052688    // [2][64][8]
#define OFF_COEF 1053712    // [2][64][8]  a2*gnw*rsig
#define OFF_PCON 1054736    // [2][64][8]  a2*(gnb - gnw*mu*rsig)

__device__ __forceinline__ float sigf(float v) { return 1.0f / (1.0f + __expf(-v)); }

// ---------------------------------------------------------------------------
// Kernel 1: per (n,c): row sums (over t), col sums (over f), 4 corners; r & i.
// ---------------------------------------------------------------------------
__global__ __launch_bounds__(256) void k1_reduce(const float* __restrict__ x,
                                                 float* __restrict__ ws) {
  const int blk = blockIdx.x;
  const int n = blk >> 3, c = blk & 7;
  const int b = n >> 3, cabs = (n & 7) * 8 + c;
  const int tid = threadIdx.x;
  const int lane = tid & 31, wv = tid >> 5;
  __shared__ float2 rowpart[FF][8];

  const float2* xp = (const float2*)x + ((size_t)b * 64 + cabs) * FF * TT;
  float cs_r = 0.f, cs_i = 0.f;
  for (int f = 0; f < FF; ++f) {
    float2 v = xp[(size_t)f * TT + tid];
    cs_r += v.x; cs_i += v.y;
    float rr = v.x, ri = v.y;
#pragma unroll
    for (int mk = 16; mk >= 1; mk >>= 1) {
      rr += __shfl_xor(rr, mk, 32);
      ri += __shfl_xor(ri, mk, 32);
    }
    if (lane == 0) rowpart[f][wv] = make_float2(rr, ri);
    if ((f == 0 || f == FF - 1) && (tid == 0 || tid == TT - 1)) {
      int ci = (f ? 2 : 0) + (tid ? 1 : 0);
      ws[OFF_CORN + ((0 * NN + n) * CG + c) * 4 + ci] = v.x;
      ws[OFF_CORN + ((1 * NN + n) * CG + c) * 4 + ci] = v.y;
    }
  }
  ws[OFF_COL + ((0 * NN + n) * CG + c) * TT + tid] = cs_r;
  ws[OFF_COL + ((1 * NN + n) * CG + c) * TT + tid] = cs_i;
  __syncthreads();
  float sr = 0.f, si = 0.f;
#pragma unroll
  for (int w = 0; w < 8; ++w) { float2 p = rowpart[tid][w]; sr += p.x; si += p.y; }
  ws[OFF_ROW + ((0 * NN + n) * CG + c) * FF + tid] = sr;
  ws[OFF_ROW + ((1 * NN + n) * CG + c) * FF + tid] = si;
}

// ---------------------------------------------------------------------------
// Kernel 2 (per n): 1x1-conv gates wf/wt; a1 = softmax(gnb) (exact, GN mean
// equals bias); a2 = softmax(mean(conv3x3)) via border-corrected sums (exact).
// ---------------------------------------------------------------------------
__global__ __launch_bounds__(256) void k2_gates(
    const float* __restrict__ w1r, const float* __restrict__ b1r,
    const float* __restrict__ w1i, const float* __restrict__ b1i,
    const float* __restrict__ w3r, const float* __restrict__ b3r,
    const float* __restrict__ w3i, const float* __restrict__ b3i,
    const float* __restrict__ gnb_r, const float* __restrict__ gnb_i,
    float* __restrict__ ws) {
  const int n = blockIdx.x;
  const int tid = threadIdx.x;
  __shared__ float Ssh[2][CG][9];
  __shared__ float tot[2][CG];
  __shared__ float m2[2][CG];

  // gates: s = tid covers the f-half (half 0) and t-half (half 1)
#pragma unroll
  for (int half = 0; half < 2; ++half) {
    const int srcBase = half ? OFF_COL : OFF_ROW;
    const int dstBase = half ? OFF_WT : OFF_WF;
    float catr[CG], cati[CG];
#pragma unroll
    for (int c = 0; c < CG; ++c) {
      catr[c] = ws[srcBase + ((0 * NN + n) * CG + c) * 256 + tid] * (1.f / 256.f);
      cati[c] = ws[srcBase + ((1 * NN + n) * CG + c) * 256 + tid] * (1.f / 256.f);
    }
#pragma unroll
    for (int o = 0; o < CG; ++o) {
      float hr = b1r[o], hi = b1i[o];
#pragma unroll
      for (int c = 0; c < CG; ++c) {
        hr += w1r[o * CG + c] * catr[c];
        hi += w1i[o * CG + c] * cati[c];
      }
      float sr = sigf(hr), si = sigf(hi);
      ws[dstBase + ((0 * NN + n) * CG + o) * 256 + tid] = sr - si;
      ws[dstBase + ((1 * NN + n) * CG + o) * 256 + tid] = si + sr;
    }
  }
  // totals
  if (tid < 16) {
    int br = tid >> 3, c = tid & 7;
    float t = 0.f;
    for (int f = 0; f < FF; ++f) t += ws[OFF_ROW + ((br * NN + n) * CG + c) * FF + f];
    tot[br][c] = t;
  }
  __syncthreads();
  // shifted sums S_c(ky,kx) with SAME zero padding (border-corrected)
  if (tid < 144) {
    int br = tid / 72, rem = tid % 72, c = rem / 9, k9 = rem % 9;
    int dy = (k9 / 3) - 1, dx = (k9 % 3) - 1;
    float S = tot[br][c];
    int rex = (dy > 0) ? 0 : FF - 1;
    int cex = (dx > 0) ? 0 : TT - 1;
    if (dy != 0) S -= ws[OFF_ROW + ((br * NN + n) * CG + c) * FF + rex];
    if (dx != 0) S -= ws[OFF_COL + ((br * NN + n) * CG + c) * TT + cex];
    if (dy != 0 && dx != 0)
      S += ws[OFF_CORN + ((br * NN + n) * CG + c) * 4 + (rex ? 2 : 0) + (cex ? 1 : 0)];
    Ssh[br][c][k9] = S;
  }
  __syncthreads();
  if (tid < 16) {
    int br = tid >> 3, o = tid & 7;
    const float* w3 = br ? w3i : w3r;
    const float* b3 = br ? b3i : b3r;
    float m = b3[o];
#pragma unroll
    for (int c = 0; c < CG; ++c)
#pragma unroll
      for (int k = 0; k < 9; ++k)
        m += w3[(o * CG + c) * 9 + k] * Ssh[br][c][k] * FTI;
    m2[br][o] = m;
  }
  __syncthreads();
  if (tid < 2) {
    int br = tid;
    float mx = -1e30f;
    for (int o = 0; o < CG; ++o) mx = fmaxf(mx, m2[br][o]);
    float e[CG], sum = 0.f;
    for (int o = 0; o < CG; ++o) { e[o] = __expf(m2[br][o] - mx); sum += e[o]; }
    for (int o = 0; o < CG; ++o) ws[OFF_A2 + (br * NN + n) * CG + o] = e[o] / sum;
    // a1 = softmax(gnb): constant across n (duplicate identical writes are benign)
    const float* gb = br ? gnb_i : gnb_r;
    float mx1 = -1e30f;
    for (int o = 0; o < CG; ++o) mx1 = fmaxf(mx1, gb[o]);
    float e1[CG], s1 = 0.f;
    for (int o = 0; o < CG; ++o) { e1[o] = __expf(gb[o] - mx1); s1 += e1[o]; }
    for (int o = 0; o < CG; ++o) ws[OFF_A1 + br * CG + o] = e1[o] / s1;
  }
}

// ---------------------------------------------------------------------------
// Kernel 3 (per n,c): mean/var of y = x*wf*wt -> coef/const for the closed-form
// a2-contracted GroupNorm term.
// ---------------------------------------------------------------------------
__global__ __launch_bounds__(256) void k3_gnstats(
    const float* __restrict__ x,
    const float* __restrict__ gnw_r, const float* __restrict__ gnb_r,
    const float* __restrict__ gnw_i, const float* __restrict__ gnb_i,
    float* __restrict__ ws) {
  const int blk = blockIdx.x;
  const int n = blk >> 3, c = blk & 7;
  const int b = n >> 3, cabs = (n & 7) * 8 + c;
  const int tid = threadIdx.x;
  const float2* xp = (const float2*)x + ((size_t)b * 64 + cabs) * FF * TT;
  const float wtr = ws[OFF_WT + ((0 * NN + n) * CG + c) * TT + tid];
  const float wti = ws[OFF_WT + ((1 * NN + n) * CG + c) * TT + tid];
  float s1r = 0.f, s2r = 0.f, s1i = 0.f, s2i = 0.f;
  for (int f = 0; f < FF; ++f) {
    float wfr = ws[OFF_WF + ((0 * NN + n) * CG + c) * FF + f];
    float wfi = ws[OFF_WF + ((1 * NN + n) * CG + c) * FF + f];
    float2 v = xp[(size_t)f * TT + tid];
    float yr = v.x * wfr * wtr, yi = v.y * wfi * wti;
    s1r += yr; s2r += yr * yr; s1i += yi; s2i += yi * yi;
  }
  __shared__ float red[8][4];
  const int lane = tid & 31, wv = tid >> 5;
#pragma unroll
  for (int mk = 16; mk >= 1; mk >>= 1) {
    s1r += __shfl_xor(s1r, mk, 32); s2r += __shfl_xor(s2r, mk, 32);
    s1i += __shfl_xor(s1i, mk, 32); s2i += __shfl_xor(s2i, mk, 32);
  }
  if (lane == 0) { red[wv][0] = s1r; red[wv][1] = s2r; red[wv][2] = s1i; red[wv][3] = s2i; }
  __syncthreads();
  if (tid == 0) {
    float a = 0.f, bb = 0.f, cc = 0.f, d = 0.f;
    for (int w = 0; w < 8; ++w) { a += red[w][0]; bb += red[w][1]; cc += red[w][2]; d += red[w][3]; }
    float mur = a * FTI, vr = bb * FTI - mur * mur;
    float mui = cc * FTI, vi = d * FTI - mui * mui;
    float rr = rsqrtf(vr + 1e-5f), ri = rsqrtf(vi + 1e-5f);
    float a2r = ws[OFF_A2 + (0 * NN + n) * CG + c];
    float a2i = ws[OFF_A2 + (1 * NN + n) * CG + c];
    ws[OFF_COEF + (0 * NN + n) * CG + c] = a2r * gnw_r[c] * rr;
    ws[OFF_COEF + (1 * NN + n) * CG + c] = a2i * gnw_i[c] * ri;
    ws[OFF_PCON + (0 * NN + n) * CG + c] = a2r * (gnb_r[c] - gnw_r[c] * mur * rr);
    ws[OFF_PCON + (1 * NN + n) * CG + c] = a2i * (gnb_i[c] - gnw_i[c] * mui * ri);
  }
}

// ---------------------------------------------------------------------------
// Kernel 4: fused conv3x3 (WMMA bf16 hi/lo split, fp32 acc, a1 folded into B)
// + closed-form a2*GN term + sigmoid gate + output. One block per (n, 32 rows).
// Branch-free LDS gathers; contracted conv weights computed cooperatively in
// LDS once (avoids per-lane redundant w3 loads in the B register build).
// ---------------------------------------------------------------------------
#define ROWS 32
#define TP 258   // padded t extent: index = t + 1, t in [-1, 256]
#define STRD 10  // floats per (t,br): 8 channels + 2 pad (80B t-stride, conflict-free)

__device__ __forceinline__ void gather8(float* dst, const float* st, int slot,
                                        int tpad, int br) {
  const float* p = st + ((slot * TP + tpad) * 2 + br) * STRD;
  float2 a = *(const float2*)(p + 0);
  float2 b = *(const float2*)(p + 2);
  float2 c = *(const float2*)(p + 4);
  float2 d = *(const float2*)(p + 6);
  dst[0] = a.x; dst[1] = a.y; dst[2] = b.x; dst[3] = b.y;
  dst[4] = c.x; dst[5] = c.y; dst[6] = d.x; dst[7] = d.y;
}

__device__ __forceinline__ void stage_row(const float* __restrict__ x, float* st,
                                          int n, int f, int slot, int tid) {
  const bool ok = (unsigned)f < (unsigned)FF;
  const int b = n >> 3, cbase = (n & 7) * 8;
#pragma unroll
  for (int k = 0; k < 4; ++k) {
    int idx = tid + 256 * k;   // 1024 float4 units: 8 c * 128
    int c = idx >> 7;
    int tq = idx & 127;        // t = 2*tq, float4 = (t:r,i)(t+1:r,i)
    float4 g = make_float4(0.f, 0.f, 0.f, 0.f);
    if (ok) {
      const float4* gp =
          (const float4*)(x + (((size_t)b * 64 + cbase + c) * FF + f) * TT * 2);
      g = gp[tq];
      if (f + 3 < FF)  // keep the row pipeline ahead: emits global_prefetch_b8
        __builtin_prefetch(
            (const void*)(x + (((size_t)b * 64 + cbase + c) * FF + (f + 3)) * TT * 2 + tq * 4),
            0, 0);
    }
    int tp = tq * 2 + 1;  // padded index
    st[((slot * TP + tp) * 2 + 0) * STRD + c] = g.x;
    st[((slot * TP + tp) * 2 + 1) * STRD + c] = g.y;
    st[((slot * TP + tp + 1) * 2 + 0) * STRD + c] = g.z;
    st[((slot * TP + tp + 1) * 2 + 1) * STRD + c] = g.w;
  }
  // zero edge columns t = -1 and t = 256 (padded 0 and 257)
  if (tid < 32) {
    int tp = (tid & 16) ? (TP - 1) : 0;
    int br = (tid >> 3) & 1;
    int c = tid & 7;
    st[((slot * TP + tp) * 2 + br) * STRD + c] = 0.f;
  }
}

__global__ __launch_bounds__(256) void k4_main(
    const float* __restrict__ x,
    const float* __restrict__ w3r, const float* __restrict__ b3r,
    const float* __restrict__ w3i, const float* __restrict__ b3i,
    const float* __restrict__ ws, float* __restrict__ dout) {
  __shared__ float st[3 * TP * 2 * STRD];   // ~60.5 KB sliding 3-row window
  __shared__ float tilebuf[8][2][16];
  __shared__ float rowco[2][CG];
  __shared__ float wcs[2][3][24];           // a1-contracted conv weights
  __shared__ float cst[2];

  const int n = blockIdx.y;
  const int f0 = blockIdx.x * ROWS;
  const int tid = threadIdx.x;
  const int lane = tid & 31, wv = tid >> 5;
  const int m = lane & 15;
  const bool hiH = lane >= 16;
  const int h = hiH ? 1 : 0;
  const int b = n >> 3, cbase = (n & 7) * 8;

  if (tid < 2) {
    int br = tid;
    const float* b3 = br ? b3i : b3r;
    float csum = 0.f;
    for (int c = 0; c < CG; ++c) csum += ws[OFF_PCON + (br * NN + n) * CG + c];
    for (int o = 0; o < CG; ++o) csum += ws[OFF_A1 + br * CG + o] * b3[o];
    cst[br] = csum;
  }
  // cooperative a1-contraction of w3: wcs[br][ky][K], K = kx*8 + c
  if (tid < 144) {
    int br = tid / 72, tap = tid % 72;
    int ky = tap / 24, K = tap % 24;
    int kx = K >> 3, c = K & 7;
    const float* w3 = br ? w3i : w3r;
    float w = 0.f;
    for (int o = 0; o < CG; ++o)
      w += ws[OFF_A1 + br * CG + o] * w3[((o * CG + c) * 3 + ky) * 3 + kx];
    wcs[br][ky][K] = w;
  }

  // loop-invariant wt gates for this lane's two tile positions
  float wtreg_r[2][4], wtreg_i[2][4];
#pragma unroll
  for (int j = 0; j < 2; ++j) {
    int t = (wv * 2 + j) * 16 + m;
#pragma unroll
    for (int c4 = 0; c4 < 4; ++c4) {
      int c = 4 * h + c4;
      wtreg_r[j][c4] = ws[OFF_WT + ((0 * NN + n) * CG + c) * TT + t];
      wtreg_i[j][c4] = ws[OFF_WT + ((1 * NN + n) * CG + c) * TT + t];
    }
  }

  // rotating slots: sP holds row f-1, sC row f, sN row f+1
  int sP = 0, sC = 1, sN = 2;
  stage_row(x, st, n, f0 - 1, sP, tid);
  stage_row(x, st, n, f0, sC, tid);
  stage_row(x, st, n, f0 + 1, sN, tid);
  if (tid < 16) {
    int br = tid >> 3, c = tid & 7;
    rowco[br][c] = ws[OFF_COEF + (br * NN + n) * CG + c] *
                   ws[OFF_WF + ((br * NN + n) * CG + c) * FF + f0];
  }
  __syncthreads();

  // B matrices from the LDS-contracted weights: only column N=0 nonzero, so
  // each WMMA's column 0 is the a1-weighted conv directly. K 24..31 zero.
  // ISA B layout (16x16x32, 16-bit): lanes 0-15 N=l hold K 0..7,16..23; lanes 16-31 K 8..15,24..31.
  v16bf Bh[2][3], Bl[2][3];
#pragma unroll
  for (int br = 0; br < 2; ++br) {
#pragma unroll
    for (int ky = 0; ky < 3; ++ky) {
#pragma unroll
      for (int e = 0; e < 16; ++e) {
        int K = (e < 8) ? (8 * h + e) : (16 + 8 * h + (e - 8));
        float w = (K < 24 && m == 0) ? wcs[br][ky][K < 24 ? K : 0] : 0.f;
        __bf16 hh = (__bf16)w;
        Bh[br][ky][e] = hh;
        Bl[br][ky][e] = (__bf16)(w - (float)hh);
      }
    }
  }

  const float cst0 = cst[0], cst1 = cst[1];

  for (int r = 0; r < ROWS; ++r) {
    const int f = f0 + r;
#pragma unroll
    for (int j = 0; j < 2; ++j) {
      const int t0 = (wv * 2 + j) * 16;
      // Uniform, branch-free A gathers. ISA A layout (16x16x32, 16-bit):
      // lanes 0-15 (M=m):  e 0..7 -> K 0..7 (kx=0, t-1); e 8..15 -> K 16..23 (kx=2, t+1)
      // lanes 16-31 (M=m): e 0..7 -> K 8..15 (kx=1, t);  e 8..15 -> K 24..31 (don't-care:
      //                    B rows K=24..31 are zero, so load any in-bounds data)
      const int tA = t0 + m + (hiH ? 1 : 0);
      const int tB = hiH ? tA : (t0 + m + 2);
#pragma unroll
      for (int br = 0; br < 2; ++br) {
        v8f C = {0.f, 0.f, 0.f, 0.f, 0.f, 0.f, 0.f, 0.f};
#pragma unroll
        for (int ky = 0; ky < 3; ++ky) {
          const int sl = (ky == 0) ? sP : ((ky == 1) ? sC : sN);
          float av[16];
          gather8(av, st, sl, tA, br);
          gather8(av + 8, st, sl, tB, br);
          v16bf ah, al;
#pragma unroll
          for (int e = 0; e < 16; ++e) {
            __bf16 hh = (__bf16)av[e];
            ah[e] = hh;
            al[e] = (__bf16)(av[e] - (float)hh);
          }
          C = __builtin_amdgcn_wmma_f32_16x16x32_bf16(false, ah, false, Bh[br][ky],
                                                      (short)0, C, false, false);
          C = __builtin_amdgcn_wmma_f32_16x16x32_bf16(false, ah, false, Bl[br][ky],
                                                      (short)0, C, false, false);
          C = __builtin_amdgcn_wmma_f32_16x16x32_bf16(false, al, false, Bh[br][ky],
                                                      (short)0, C, false, false);
        }
        // column 0 of C = a1-contracted conv for positions M=v(+8 for hi half)
        if (m == 0) {
#pragma unroll
          for (int v = 0; v < 8; ++v) tilebuf[wv][br][(hiH ? 8 : 0) + v] = C[v];
        }
      }
      // output phase: lane -> (position m, channel-half h); same-wave LDS is in-order
      {
        const int t = t0 + m;
        float convr = tilebuf[wv][0][m];
        float convi = tilebuf[wv][1][m];
        const float* pr = st + ((sC * TP + t + 1) * 2 + 0) * STRD + 4 * h;
        const float* pi = st + ((sC * TP + t + 1) * 2 + 1) * STRD + 4 * h;
        float xr[4], xi[4];
        float2 q;
        q = *(const float2*)(pr + 0); xr[0] = q.x; xr[1] = q.y;
        q = *(const float2*)(pr + 2); xr[2] = q.x; xr[3] = q.y;
        q = *(const float2*)(pi + 0); xi[0] = q.x; xi[1] = q.y;
        q = *(const float2*)(pi + 2); xi[2] = q.x; xi[3] = q.y;
        float prs = 0.f, pis = 0.f;
#pragma unroll
        for (int c4 = 0; c4 < 4; ++c4) {
          int c = 4 * h + c4;
          prs += rowco[0][c] * wtreg_r[j][c4] * xr[c4];
          pis += rowco[1][c] * wtreg_i[j][c4] * xi[c4];
        }
        prs += __shfl_xor(prs, 16, 32);
        pis += __shfl_xor(pis, 16, 32);
        float gr = sigf(cst0 + convr + prs);
        float gi = sigf(cst1 + convi + pis);
#pragma unroll
        for (int c4 = 0; c4 < 4; ++c4) {
          int c = 4 * h + c4;
          size_t oidx = (((size_t)b * 64 + cbase + c) * FF + f) * TT + t;
          ((float2*)dout)[oidx] = make_float2(xr[c4] * gr, xi[c4] * gi);
        }
      }
    }
    __syncthreads();
    if (r < ROWS - 1) {
      stage_row(x, st, n, f + 2, sP, tid);  // row f+2 replaces row f-1
      if (tid < 16) {
        int br = tid >> 3, c = tid & 7;
        rowco[br][c] = ws[OFF_COEF + (br * NN + n) * CG + c] *
                       ws[OFF_WF + ((br * NN + n) * CG + c) * FF + (f + 1)];
      }
      // rotate slots
      int tmp = sP; sP = sC; sC = sN; sN = tmp;
      __syncthreads();
    }
  }
}

// ---------------------------------------------------------------------------
extern "C" void kernel_launch(void* const* d_in, const int* in_sizes, int n_in,
                              void* d_out, int out_size, void* d_ws, size_t ws_size,
                              hipStream_t stream) {
  const float* x     = (const float*)d_in[0];
  const float* w1r   = (const float*)d_in[1];
  const float* b1r   = (const float*)d_in[2];
  const float* w1i   = (const float*)d_in[3];
  const float* b1i   = (const float*)d_in[4];
  const float* w3r   = (const float*)d_in[5];
  const float* b3r   = (const float*)d_in[6];
  const float* w3i   = (const float*)d_in[7];
  const float* b3i   = (const float*)d_in[8];
  const float* gnw_r = (const float*)d_in[9];
  const float* gnb_r = (const float*)d_in[10];
  const float* gnw_i = (const float*)d_in[11];
  const float* gnb_i = (const float*)d_in[12];
  float* ws  = (float*)d_ws;   // needs ~4.3 MB
  float* out = (float*)d_out;

  k1_reduce<<<512, 256, 0, stream>>>(x, ws);
  k2_gates<<<64, 256, 0, stream>>>(w1r, b1r, w1i, b1i, w3r, b3r, w3i, b3i,
                                   gnb_r, gnb_i, ws);
  k3_gnstats<<<512, 256, 0, stream>>>(x, gnw_r, gnb_r, gnw_i, gnb_i, ws);
  k4_main<<<dim3(FF / ROWS, NN), 256, 0, stream>>>(x, w3r, b3r, w3i, b3i, ws, out);
}
